// HeteroDLSTM_67697274520451
// MI455X (gfx1250) — compile-verified, matrix operand order checked by reading, not run on hardware
//
#include <hip/hip_runtime.h>
#include <hip/hip_bf16.h>
#include <math.h>

// ---------------------------------------------------------------------------
// Types for CDNA5 WMMA
// ---------------------------------------------------------------------------
typedef __attribute__((ext_vector_type(16))) __bf16          v16bf;
typedef __attribute__((ext_vector_type(8)))  float           v8f;

struct Frag32 { uint4 lo; uint4 hi; };   // 32 bytes = 16 bf16

__device__ __forceinline__ unsigned short f2bf(float f) {
  unsigned u = __float_as_uint(f);
  u += 0x7fffu + ((u >> 16) & 1u);   // round-to-nearest-even
  return (unsigned short)(u >> 16);
}
// pack two fp32 -> packed bf16x2 (prefer hardware v_cvt_pk_bf16_f32)
__device__ __forceinline__ unsigned pk2(float a, float b) {
#if __has_builtin(__builtin_amdgcn_cvt_pk_bf16_f32)
  typedef __attribute__((ext_vector_type(2))) __bf16 v2bf;
  v2bf r = __builtin_amdgcn_cvt_pk_bf16_f32(a, b);
  return __builtin_bit_cast(unsigned, r);
#else
  return (unsigned)f2bf(a) | ((unsigned)f2bf(b) << 16);
#endif
}

// order-preserving float<->uint for atomic scatter-max
__device__ __forceinline__ unsigned f2u_ord(float f) {
  unsigned u = __float_as_uint(f);
  return (u & 0x80000000u) ? ~u : (u | 0x80000000u);
}
__device__ __forceinline__ float u2f_ord(unsigned u) {
  unsigned v = (u & 0x80000000u) ? (u & 0x7fffffffu) : ~u;
  return __uint_as_float(v);
}
#define NEG_INF_ENC 0x007FFFFFu   // f2u_ord(-inf)

__device__ __forceinline__ float dec_agg(unsigned e) {
  if (e == NEG_INF_ENC) return 0.0f;   // empty segment -> 0 (PyG MaxAggregation)
  return u2f_ord(e);
}

__device__ __forceinline__ float sigf(float x) { return 1.0f / (1.0f + expf(-x)); }

// ---------------------------------------------------------------------------
// bf16 WMMA GEMM:  C[M,N] = A[M,K] * B  (+ bias | += C)
//   TRANSB == false : B is [K,N] row-major
//   TRANSB == true  : B is [N,K] row-major (computes A * B^T)
//
// Block: 256 threads = 8 waves; tile 64(M) x 128(N) x 32(K).
// Both A and B tiles live in LDS in "fragment-canonical" [row][k] layout
// (B stored transposed) so every WMMA operand is two ds_load_b128's.
// Requires: K % 32 == 0, N % 128 == 0, M % 16 == 0 (all hold here).
// ---------------------------------------------------------------------------
#define BM 64
#define BN 128
#define BKK 32
#define LDT 40   // padded row stride in ushorts (80B, 16B-aligned chunks)

__device__ __forceinline__ v16bf frag_ld(const unsigned short* row, int kb) {
  Frag32 f;
  f.lo = *(const uint4*)(row + kb);        // K[kb .. kb+8)
  f.hi = *(const uint4*)(row + kb + 16);   // K[kb+16 .. kb+24)
  return __builtin_bit_cast(v16bf, f);
}

template <bool TRANSB>
__global__ __launch_bounds__(256) void gemm_bf16_wmma(
    const float* __restrict__ A, const float* __restrict__ B,
    float* __restrict__ C, int M, int N, int K,
    const float* __restrict__ bias, int accumulate) {
  __shared__ __align__(16) unsigned short As[BM][LDT];
  __shared__ __align__(16) unsigned short Bs[BN][LDT];   // transposed: [n][k]

  const int tid  = threadIdx.x;
  const int lane = tid & 31;
  const int wave = tid >> 5;
  const int m0 = blockIdx.y * BM;
  const int n0 = blockIdx.x * BN;
  const int mrow  = (wave & 3) * 16;   // wave M-strip in tile
  const int nbase = (wave >> 2) * 64;  // wave N-strip (4 subtiles of 16)
  const int kb    = (lane >> 4) * 8;   // fragment K chunk select
  const int lrow  = lane & 15;

  v8f acc[4];
  #pragma unroll
  for (int s = 0; s < 4; ++s) acc[s] = (v8f){};

  for (int k0 = 0; k0 < K; k0 += BKK) {
    // ---- stage A tile: 64 rows x 32 k, 8 floats/thread, one branch ----
    {
      int r  = tid >> 2;            // 0..63
      int cb = (tid & 3) * 8;       // 0,8,16,24
      int gr = m0 + r;
      uint4 pk = {0u, 0u, 0u, 0u};
      if (gr < M) {
        const float4* ap = (const float4*)(A + (size_t)gr * K + k0 + cb);
        float4 a0 = ap[0], a1 = ap[1];
        pk.x = pk2(a0.x, a0.y); pk.y = pk2(a0.z, a0.w);
        pk.z = pk2(a1.x, a1.y); pk.w = pk2(a1.z, a1.w);
      }
      *(uint4*)&As[r][cb] = pk;
    }
    // ---- stage B tile: 128 n-rows x 32 k (transposed layout) ----
    if (TRANSB) {
      // B[n][k] rows are contiguous: pure wide copy
      int nl = tid >> 1;            // 0..127
      int cb = (tid & 1) * 16;      // 0,16
      const float4* bp = (const float4*)(B + (size_t)(n0 + nl) * K + k0 + cb);
      float4 b0 = bp[0], b1 = bp[1], b2 = bp[2], b3 = bp[3];
      uint4 p0 = {pk2(b0.x, b0.y), pk2(b0.z, b0.w), pk2(b1.x, b1.y), pk2(b1.z, b1.w)};
      uint4 p1 = {pk2(b2.x, b2.y), pk2(b2.z, b2.w), pk2(b3.x, b3.y), pk2(b3.z, b3.w)};
      *(uint4*)&Bs[nl][cb]     = p0;
      *(uint4*)&Bs[nl][cb + 8] = p1;
    } else {
      // B[k][n] row-major: coalesced read along n, transpose into LDS
      int r  = tid >> 3;            // k row 0..31
      int nb = (tid & 7) * 16;      // 16 n's per thread
      const float4* bp = (const float4*)(B + (size_t)(k0 + r) * N + n0 + nb);
      #pragma unroll
      for (int q = 0; q < 4; ++q) {
        float4 b = bp[q];
        Bs[nb + q * 4 + 0][r] = f2bf(b.x);
        Bs[nb + q * 4 + 1][r] = f2bf(b.y);
        Bs[nb + q * 4 + 2][r] = f2bf(b.z);
        Bs[nb + q * 4 + 3][r] = f2bf(b.w);
      }
    }
    __syncthreads();

    // ---- 1 A fragment + 4 B fragments -> 4 WMMA ----
    v16bf af = frag_ld(&As[mrow + lrow][0], kb);
    #pragma unroll
    for (int s = 0; s < 4; ++s) {
      v16bf bf = frag_ld(&Bs[nbase + s * 16 + lrow][0], kb);
      acc[s] = __builtin_amdgcn_wmma_f32_16x16x32_bf16(false, af, false, bf,
                                                       (short)0, acc[s], false, false);
    }
    __syncthreads();
  }

  // ---- epilogue: strip-level row guard only (M % 16 == 0) ----
  if (m0 + mrow < M) {
    const int rbase = m0 + mrow + ((lane < 16) ? 0 : 8);
    #pragma unroll
    for (int s = 0; s < 4; ++s) {
      int c = n0 + nbase + s * 16 + lrow;
      float* cp = C + (size_t)rbase * N + c;
      float bv = (!accumulate && bias) ? bias[c] : 0.0f;
      #pragma unroll
      for (int i = 0; i < 8; ++i) {
        float v = acc[s][i];
        if (accumulate) *cp += v;
        else            *cp = v + bv;
        cp += N;
      }
    }
  }
}

// ---------------------------------------------------------------------------
// Attention epilogue: per (node, head) -> leaky_relu(xh.q), softmax over the
// 5 splits, weighted sum -> x0_spot[n, h*128 + k]
// ---------------------------------------------------------------------------
__global__ __launch_bounds__(128) void attn_epilogue(
    const float* __restrict__ xh, const float* __restrict__ q,
    float* __restrict__ x0s, int NS) {
  const int n = blockIdx.x;
  const int h = blockIdx.y;
  const int k = threadIdx.x;  // 0..127
  const float* base = xh + (size_t)h * (size_t)NS * 5 * 128;

  __shared__ float red[5][128];
  __shared__ float w[5];

  float qv = q[h * 128 + k];
  float xv[5];
  #pragma unroll
  for (int s = 0; s < 5; ++s) {
    xv[s] = base[(size_t)(n * 5 + s) * 128 + k];
    red[s][k] = xv[s] * qv;
  }
  __syncthreads();
  for (int off = 64; off > 0; off >>= 1) {
    if (k < off) {
      #pragma unroll
      for (int s = 0; s < 5; ++s) red[s][k] += red[s][k + off];
    }
    __syncthreads();
  }
  if (k == 0) {
    float sc[5], mx = -3.4e38f;
    #pragma unroll
    for (int s = 0; s < 5; ++s) {
      float a = red[s][0];
      a = (a >= 0.0f) ? a : 0.2f * a;   // leaky_relu 0.2
      sc[s] = a;
      mx = fmaxf(mx, a);
    }
    float sum = 0.0f;
    #pragma unroll
    for (int s = 0; s < 5; ++s) { sc[s] = expf(sc[s] - mx); sum += sc[s]; }
    #pragma unroll
    for (int s = 0; s < 5; ++s) w[s] = sc[s] / sum;
  }
  __syncthreads();
  float o = 0.0f;
  #pragma unroll
  for (int s = 0; s < 5; ++s) o += w[s] * xv[s];
  x0s[(size_t)n * 512 + h * 128 + k] = o;
}

// ---------------------------------------------------------------------------
// Elementwise / graph kernels
// ---------------------------------------------------------------------------
__global__ void fill_u32_k(unsigned* __restrict__ p, unsigned v, size_t n) {
  size_t i = (size_t)blockIdx.x * 256 + threadIdx.x;
  if (i < n) p[i] = v;
}
__global__ void zero_f32_k(float* __restrict__ p, size_t n) {
  size_t i = (size_t)blockIdx.x * 256 + threadIdx.x;
  if (i < n) p[i] = 0.0f;
}

// one block per edge (y = channel half); atomic scatter-max into encoded buf
__global__ __launch_bounds__(128) void scatter_max_k(
    const float* __restrict__ sp, const int* __restrict__ src,
    const int* __restrict__ dst, unsigned* __restrict__ agg) {
  int e = blockIdx.x;
  int c = blockIdx.y * 128 + threadIdx.x;
  int s = src[e];
  int d = dst[e];
  float v = sp[(size_t)s * 256 + c];
  atomicMax(&agg[(size_t)d * 256 + c], f2u_ord(v));
}

// decode two spot-targeted aggregates in place, emit meta = (a0+a2)/2
__global__ void decode2_meta_k(unsigned* __restrict__ a0, unsigned* __restrict__ a2,
                               float* __restrict__ meta, size_t n) {
  size_t i = (size_t)blockIdx.x * 256 + threadIdx.x;
  if (i >= n) return;
  float v0 = dec_agg(a0[i]);
  float v2 = dec_agg(a2[i]);
  a0[i] = __float_as_uint(v0);
  a2[i] = __float_as_uint(v2);
  meta[i] = 0.5f * (v0 + v2);
}
__global__ void decode1_meta_k(unsigned* __restrict__ a1, float* __restrict__ meta,
                               size_t n) {
  size_t i = (size_t)blockIdx.x * 256 + threadIdx.x;
  if (i >= n) return;
  float v = dec_agg(a1[i]);
  a1[i] = __float_as_uint(v);
  meta[i] = v;
}

// LSTMCell gates (i,f,g,o quarters of 1024-wide g) -> acc += h
__global__ __launch_bounds__(128) void lstm_gate_k(
    const float* __restrict__ g, const float* __restrict__ meta,
    float* __restrict__ acc) {
  int n = blockIdx.x;
  int k = blockIdx.y * 128 + threadIdx.x;
  const float* gr = g + (size_t)n * 1024;
  float gi = gr[k], gf = gr[256 + k], gg = gr[512 + k], go = gr[768 + k];
  float c  = meta[(size_t)n * 256 + k];
  float c2 = sigf(gf) * c + sigf(gi) * tanhf(gg);
  acc[(size_t)n * 256 + k] += sigf(go) * tanhf(c2);
}

__global__ void relu_scale_k(const float* __restrict__ a, float* __restrict__ o,
                             float scale, size_t n) {
  size_t i = (size_t)blockIdx.x * 256 + threadIdx.x;
  if (i < n) o[i] = fmaxf(a[i] * scale, 0.0f);
}

__global__ void avg_half_k(const float* __restrict__ a, const float* __restrict__ b,
                           float* __restrict__ o, size_t n) {
  size_t i = (size_t)blockIdx.x * 256 + threadIdx.x;
  if (i < n) o[i] = 0.5f * (a[i] + b[i]);
}

// out[n] = dot(X[n,:256], w) + b[0]
__global__ __launch_bounds__(256) void rowdot_k(
    const float* __restrict__ X, const float* __restrict__ w,
    const float* __restrict__ b, float* __restrict__ out) {
  int n = blockIdx.x;
  int t = threadIdx.x;
  __shared__ float red[256];
  red[t] = X[(size_t)n * 256 + t] * w[t];
  __syncthreads();
  for (int off = 128; off > 0; off >>= 1) {
    if (t < off) red[t] += red[t + off];
    __syncthreads();
  }
  if (t == 0) out[n] = red[0] + b[0];
}

// ---------------------------------------------------------------------------
// Launcher
// ---------------------------------------------------------------------------
extern "C" void kernel_launch(void* const* d_in, const int* in_sizes, int n_in,
                              void* d_out, int out_size, void* d_ws, size_t ws_size,
                              hipStream_t stream) {
  const float* x_spot = (const float*)d_in[0];
  const float* x_user = (const float*)d_in[1];
  const float* att_W  = (const float*)d_in[2];
  const float* att_q  = (const float*)d_in[3];
  const float* l0_Wsrc[3] = {(const float*)d_in[4], (const float*)d_in[5], (const float*)d_in[6]};
  const float* l0_Wtgt[3] = {(const float*)d_in[7], (const float*)d_in[8], (const float*)d_in[9]};
  const float* l0_Wih = (const float*)d_in[10];
  const float* l0_Whh = (const float*)d_in[11];
  const float* l0_b   = (const float*)d_in[12];
  const float* l1_Wsrc = (const float*)d_in[13];
  const float* l1_Wtgt = (const float*)d_in[14];
  const float* l1_Wih  = (const float*)d_in[15];
  const float* l1_Whh  = (const float*)d_in[16];
  const float* l1_b    = (const float*)d_in[17];
  const float* lin_sW = (const float*)d_in[18];
  const float* lin_sb = (const float*)d_in[19];
  const float* lin_uW = (const float*)d_in[20];
  const float* lin_ub = (const float*)d_in[21];
  const int* src_us = (const int*)d_in[22];
  const int* dst_us = (const int*)d_in[23];
  const int* src_su = (const int*)d_in[24];
  const int* dst_su = (const int*)d_in[25];
  const int* src_ss = (const int*)d_in[26];
  const int* dst_ss = (const int*)d_in[27];

  const int NS = in_sizes[0] / 2560;   // 20000
  const int NU = in_sizes[1] / 256;    // 40000
  const int E_us = in_sizes[22], E_su = in_sizes[24], E_ss = in_sizes[26];
  const size_t NSC = (size_t)NS * 256, NUC = (size_t)NU * 256;

  // ---- workspace layout (fp32 elements) ----
  float* ws = (float*)d_ws;
  size_t XH  = 0;                               // 4 heads * NS*5*128 (attention scratch)
  size_t GB  = XH;                              // gate buffer aliases XH (XH dead by then)
  size_t X0S = XH + (size_t)4 * NS * 5 * 128;   // NS*512
  size_t SP0 = X0S + (size_t)NS * 512;          // NU*256  (user->spot src proj)
  size_t SP1 = SP0 + NUC;                       // NS*256
  size_t SP2 = SP1 + NSC;                       // NS*256
  size_t TP0 = SP2 + NSC;                       // NS*256
  size_t TP1 = TP0 + NSC;                       // NU*256
  size_t TP2 = TP1 + NUC;                       // NS*256
  size_t AG0 = TP2 + NSC;                       // NS*256 (enc/dec in place)
  size_t AG1 = AG0 + NSC;                       // NU*256
  size_t AG2 = AG1 + NUC;                       // NS*256
  size_t MS  = AG2 + NSC;                       // NS*256
  size_t MU  = MS + NSC;                        // NU*256
  size_t ACS = MU + NUC;                        // NS*256
  size_t ACU = ACS + NSC;                       // NU*256
  size_t X1S = ACU + NUC;
  size_t X1U = X1S + NSC;
  size_t X2S = X1U + NUC;
  size_t X2U = X2S + NSC;

  auto gemmN = [&](const float* A, const float* B, float* C, int M, int N, int K,
                   const float* bias, int accum) {
    dim3 g((N + BN - 1) / BN, (M + BM - 1) / BM);
    gemm_bf16_wmma<false><<<g, 256, 0, stream>>>(A, B, C, M, N, K, bias, accum);
  };
  auto gemmT = [&](const float* A, const float* B, float* C, int M, int N, int K,
                   const float* bias, int accum) {
    dim3 g((N + BN - 1) / BN, (M + BM - 1) / BM);
    gemm_bf16_wmma<true><<<g, 256, 0, stream>>>(A, B, C, M, N, K, bias, accum);
  };
  auto blocks = [](size_t n) { return (unsigned)((n + 255) / 256); };

  // ---- 1) attention: xh[h] = x_spot(as [NS*5,512]) @ W[h]  (WMMA) ----
  for (int h = 0; h < 4; ++h)
    gemmN(x_spot, att_W + (size_t)h * 512 * 128, ws + XH + (size_t)h * NS * 5 * 128,
          NS * 5, 128, 512, nullptr, 0);
  attn_epilogue<<<dim3(NS, 4), 128, 0, stream>>>(ws + XH, att_q, ws + X0S, NS);

  // ---- 2) one hetero layer ----
  auto run_layer = [&](const float* xs, int ds, const float* xu, int du,
                       const float* Ws0, const float* Ws1, const float* Ws2,
                       const float* Wt0, const float* Wt1, const float* Wt2,
                       const float* Wih, const float* Whh, const float* bb,
                       float* xos, float* xou) {
    // source / target projections
    gemmN(xu, Ws0, ws + SP0, NU, 256, du, nullptr, 0);   // user src (us)
    gemmN(xs, Ws1, ws + SP1, NS, 256, ds, nullptr, 0);   // spot src (su)
    gemmN(xs, Ws2, ws + SP2, NS, 256, ds, nullptr, 0);   // spot src (ss)
    gemmN(xs, Wt0, ws + TP0, NS, 256, ds, nullptr, 0);   // spot tgt (us)
    gemmN(xu, Wt1, ws + TP1, NU, 256, du, nullptr, 0);   // user tgt (su)
    gemmN(xs, Wt2, ws + TP2, NS, 256, ds, nullptr, 0);   // spot tgt (ss)

    unsigned* ag0 = (unsigned*)(ws + AG0);
    unsigned* ag1 = (unsigned*)(ws + AG1);
    unsigned* ag2 = (unsigned*)(ws + AG2);
    fill_u32_k<<<blocks(NSC), 256, 0, stream>>>(ag0, NEG_INF_ENC, NSC);
    fill_u32_k<<<blocks(NUC), 256, 0, stream>>>(ag1, NEG_INF_ENC, NUC);
    fill_u32_k<<<blocks(NSC), 256, 0, stream>>>(ag2, NEG_INF_ENC, NSC);
    scatter_max_k<<<dim3(E_us, 2), 128, 0, stream>>>(ws + SP0, src_us, dst_us, ag0);
    scatter_max_k<<<dim3(E_su, 2), 128, 0, stream>>>(ws + SP1, src_su, dst_su, ag1);
    scatter_max_k<<<dim3(E_ss, 2), 128, 0, stream>>>(ws + SP2, src_ss, dst_ss, ag2);
    decode2_meta_k<<<blocks(NSC), 256, 0, stream>>>(ag0, ag2, ws + MS, NSC);
    decode1_meta_k<<<blocks(NUC), 256, 0, stream>>>(ag1, ws + MU, NUC);

    zero_f32_k<<<blocks(NSC), 256, 0, stream>>>(ws + ACS, NSC);
    zero_f32_k<<<blocks(NUC), 256, 0, stream>>>(ws + ACU, NUC);

    // edge type 0: target spot
    gemmT(ws + TP0, Wih + 0 * 1024 * 256, ws + GB, NS, 1024, 256, bb + 0 * 1024, 0);
    gemmT(ws + AG0, Whh + 0 * 1024 * 256, ws + GB, NS, 1024, 256, nullptr, 1);
    lstm_gate_k<<<dim3(NS, 2), 128, 0, stream>>>(ws + GB, ws + MS, ws + ACS);
    // edge type 1: target user
    gemmT(ws + TP1, Wih + 1 * 1024 * 256, ws + GB, NU, 1024, 256, bb + 1 * 1024, 0);
    gemmT(ws + AG1, Whh + 1 * 1024 * 256, ws + GB, NU, 1024, 256, nullptr, 1);
    lstm_gate_k<<<dim3(NU, 2), 128, 0, stream>>>(ws + GB, ws + MU, ws + ACU);
    // edge type 2: target spot
    gemmT(ws + TP2, Wih + 2 * 1024 * 256, ws + GB, NS, 1024, 256, bb + 2 * 1024, 0);
    gemmT(ws + AG2, Whh + 2 * 1024 * 256, ws + GB, NS, 1024, 256, nullptr, 1);
    lstm_gate_k<<<dim3(NS, 2), 128, 0, stream>>>(ws + GB, ws + MS, ws + ACS);

    relu_scale_k<<<blocks(NSC), 256, 0, stream>>>(ws + ACS, xos, 0.5f, NSC); // /2 edge types
    relu_scale_k<<<blocks(NUC), 256, 0, stream>>>(ws + ACU, xou, 1.0f, NUC);
  };

  run_layer(ws + X0S, 512, x_user, 256,
            l0_Wsrc[0], l0_Wsrc[1], l0_Wsrc[2],
            l0_Wtgt[0], l0_Wtgt[1], l0_Wtgt[2],
            l0_Wih, l0_Whh, l0_b, ws + X1S, ws + X1U);

  const size_t W66 = (size_t)256 * 256;
  run_layer(ws + X1S, 256, ws + X1U, 256,
            l1_Wsrc + 0 * W66, l1_Wsrc + 1 * W66, l1_Wsrc + 2 * W66,
            l1_Wtgt + 0 * W66, l1_Wtgt + 1 * W66, l1_Wtgt + 2 * W66,
            l1_Wih, l1_Whh, l1_b, ws + X2S, ws + X2U);

  // ---- 3) outputs: [avg_spot | avg_user | head_spot | head_user] ----
  float* out = (float*)d_out;
  size_t O_AS = 0, O_AU = NSC, O_HS = NSC + NUC, O_HU = O_HS + (size_t)NS;
  avg_half_k<<<blocks(NSC), 256, 0, stream>>>(ws + X1S, ws + X2S, out + O_AS, NSC);
  avg_half_k<<<blocks(NUC), 256, 0, stream>>>(ws + X1U, ws + X2U, out + O_AU, NUC);
  rowdot_k<<<NS, 256, 0, stream>>>(ws + X2S, lin_sW, lin_sb, out + O_HS);
  rowdot_k<<<NU, 256, 0, stream>>>(ws + X2U, lin_uW, lin_ub, out + O_HU);
}
